// CommRound_69569880261245
// MI455X (gfx1250) — matrix operand order repaired and checked by reading, not compiled, not run on hardware
//
#include <hip/hip_runtime.h>
#include <hip/hip_bf16.h>

// ---------------------------------------------------------------------------
// CommRound on MI455X (gfx1250): bf16 WMMA pipeline + async-to-LDS staging.
//   N=8192 agents, HIDDEN=256, MSG=128, KEY=64
// All GEMMs use v_wmma_f32_16x16x32_bf16 (f32 accumulate).
// Softmax needs no running max: scores are clipped to [-20,20].
// Attention is split 4-ways over columns (2048 waves) for occupancy;
// partial attended/rowsum are combined deterministically (no atomics).
// ---------------------------------------------------------------------------

#define NA     8192
#define HID    256
#define MSGD   128
#define KEYD   64
#define NCHUNK 4
#define JCHUNK (NA / NCHUNK)

typedef unsigned short u16;
typedef __attribute__((ext_vector_type(16))) __bf16 v16bf;
typedef __attribute__((ext_vector_type(8)))  float  v8f;
typedef __attribute__((ext_vector_type(4)))  int    v4i;
typedef __attribute__((address_space(1))) v4i* g4ptr;   // global int4*
typedef __attribute__((address_space(3))) v4i* l4ptr;   // LDS int4*
typedef __attribute__((address_space(3))) void lds_void_t;

struct B256 { uint4 lo, hi; };

static __device__ __forceinline__ u16 f2bf(float f) {
  unsigned u = __builtin_bit_cast(unsigned, f);
  u += 0x7fffu + ((u >> 16) & 1u);          // round-to-nearest-even
  return (u16)(u >> 16);
}

// A-fragment (16x32 bf16, M x K): lane ln holds row ln; K = hs*16 + e (contiguous)
static __device__ __forceinline__ v16bf load_afrag(const u16* rowPtr, int k0, int hs) {
  const uint4* p = reinterpret_cast<const uint4*>(rowPtr + k0 + hs * 16);
  B256 u; u.lo = p[0]; u.hi = p[1];         // +8 elements = +1 uint4
  return __builtin_bit_cast(v16bf, u);
}
// B-fragment (32x16 bf16, K x N): lane ln holds column n0+ln; K = hs*16 + e (contiguous)
static __device__ __forceinline__ v16bf load_bfrag(const u16* ptr) {
  const uint4* p = reinterpret_cast<const uint4*>(ptr);
  B256 u; u.lo = p[0]; u.hi = p[1];
  return __builtin_bit_cast(v16bf, u);
}
static __device__ __forceinline__ v8f wmma_bf16(v16bf a, v16bf b, v8f c) {
  return __builtin_amdgcn_wmma_f32_16x16x32_bf16(false, a, false, b, (short)0, c, false, false);
}

// CDNA5 async copy: 16B per lane, global -> LDS, tracked by ASYNCcnt.
static __device__ __forceinline__ void async_ld_b128(u16* lds_p, const u16* gptr) {
#if __has_builtin(__builtin_amdgcn_global_load_async_to_lds_b128)
  __builtin_amdgcn_global_load_async_to_lds_b128(
      (g4ptr)(void*)const_cast<u16*>(gptr), (l4ptr)(void*)lds_p, 0, 0);
#else
  unsigned la = (unsigned)(size_t)(lds_void_t*)(void*)lds_p;
  unsigned long long ga = (unsigned long long)gptr;
  asm volatile("global_load_async_to_lds_b128 %0, %1, off" :: "v"(la), "v"(ga) : "memory");
#endif
}
static __device__ __forceinline__ void wait_async0() {
#if __has_builtin(__builtin_amdgcn_s_wait_asynccnt)
  __builtin_amdgcn_s_wait_asynccnt(0);
#else
  asm volatile("s_wait_asynccnt 0x0" ::: "memory");
#endif
}

// ---------------------------------------------------------------------------
// K0a: f32 -> bf16 convert (h)
__global__ __launch_bounds__(256) void cvt_kernel(const float* __restrict__ in,
                                                  u16* __restrict__ out, int n) {
  int i = blockIdx.x * 256 + threadIdx.x;
  if (i < n) out[i] = f2bf(in[i]);
}

// K0b: transpose + convert: in[R,C] row-major f32 -> out[C,R] bf16
__global__ __launch_bounds__(256) void tcvt_kernel(const float* __restrict__ in,
                                                   u16* __restrict__ out, int R, int C) {
  int i = blockIdx.x * 256 + threadIdx.x;
  if (i < R * C) {
    int r = i / C, c = i % C;
    out[c * R + r] = f2bf(in[i]);
  }
}

// ---------------------------------------------------------------------------
// K1: projections  [msg|key|qry] = h @ [W_msg|W_key|W_qry] + bias
__global__ __launch_bounds__(256) void proj_kernel(
    const u16* __restrict__ hb,
    const u16* __restrict__ WmsgT, const u16* __restrict__ WkeyT, const u16* __restrict__ WqryT,
    const float* __restrict__ b_msg, const float* __restrict__ b_key, const float* __restrict__ b_qry,
    u16* __restrict__ MsgT, u16* __restrict__ keysB, u16* __restrict__ qryB) {
  int g = blockIdx.x * 8 + (threadIdx.x >> 5);
  int lane = threadIdx.x & 31, ln = lane & 15, hs = lane >> 4;
  int rowTile = g >> 4, colTile = g & 15;
  int rb = rowTile * 16, n0 = colTile * 16;

  const u16* WT; const float* bias; int cL;
  if (colTile < 8)       { WT = WmsgT; bias = b_msg; cL = n0; }
  else if (colTile < 12) { WT = WkeyT; bias = b_key; cL = n0 - 128; }
  else                   { WT = WqryT; bias = b_qry; cL = n0 - 192; }

  v8f acc = {0.f,0.f,0.f,0.f,0.f,0.f,0.f,0.f};
  const u16* arow = hb + (size_t)(rb + ln) * HID;
  const u16* brow = WT + (size_t)(cL + ln) * HID;
#pragma unroll
  for (int k0 = 0; k0 < HID; k0 += 32) {
    v16bf a = load_afrag(arow, k0, hs);
    v16bf b = load_bfrag(brow + k0 + hs * 16);
    acc = wmma_bf16(a, b, acc);
  }
  int c = cL + ln;
  float bv = bias[c];
#pragma unroll
  for (int v = 0; v < 8; ++v) {
    int r = rb + v + 8 * hs;
    float val = acc[v] + bv;
    if (colTile < 8)       MsgT[(size_t)c * NA + r]   = f2bf(val);   // messages transposed
    else if (colTile < 12) keysB[(size_t)r * KEYD + c] = f2bf(val);
    else                   qryB[(size_t)r * KEYD + c]  = f2bf(val);
  }
}

// ---------------------------------------------------------------------------
// K2: attention. Block = 8 waves, each owning a 16-row query block; all waves
// stream the same 2048-column chunk (blockIdx.y). Key/message tiles are staged
// cooperatively via GLOBAL_LOAD_ASYNC_TO_LDS_B128 (double-buffered, ASYNCcnt).
// Per step: QK^T (WMMA) -> exp -> unnormalized attn store + LDS relayout ->
// P@M (WMMA). Partial attended + rowsum go to workspace (combined in K2b).
__global__ __launch_bounds__(256) void attn_kernel(
    const u16* __restrict__ qryB, const u16* __restrict__ keysB, const u16* __restrict__ MsgT,
    float* __restrict__ attn, float* __restrict__ attP, float* __restrict__ rowsumP) {
  __shared__ u16 s_keys[2][32 * 72];   // 32 rows x 64, padded to 72 u16 (144 B)
  __shared__ u16 s_msg [2][128 * 40];  // 128 rows x 32, padded to 40 u16 (80 B)
  __shared__ u16 s_P   [8][16 * 40];   // per-wave 16x32 P tile, padded rows

  int tid = threadIdx.x;
  int w = tid >> 5, lane = tid & 31, ln = lane & 15, hs = lane >> 4;
  int rb = (blockIdx.x * 8 + w) * 16;
  int chunk = blockIdx.y;
  int jbeg = chunk * JCHUNK, jend = jbeg + JCHUNK;

  // cooperative stage of one (keys, msg) tile pair into buffer `buf`
  auto stage = [&](int buf, int j0) {
    {
      int row = tid >> 3, q = tid & 7;            // 32x8 chunks of 16B
      async_ld_b128(&s_keys[buf][row * 72 + q * 8],
                    keysB + (size_t)(j0 + row) * KEYD + q * 8);
    }
#pragma unroll
    for (int it = 0; it < 2; ++it) {              // 128x4 chunks of 16B
      int idx = tid + 256 * it;
      int row = idx >> 2, q = idx & 3;
      async_ld_b128(&s_msg[buf][row * 40 + q * 8],
                    MsgT + (size_t)row * NA + j0 + q * 8);
    }
  };

  v16bf qf0 = load_afrag(qryB + (size_t)(rb + ln) * KEYD, 0, hs);
  v16bf qf1 = load_afrag(qryB + (size_t)(rb + ln) * KEYD, 32, hs);

  v8f accA[8];
#pragma unroll
  for (int nt = 0; nt < 8; ++nt) accA[nt] = (v8f){0.f,0.f,0.f,0.f,0.f,0.f,0.f,0.f};
  float rs[8] = {0.f,0.f,0.f,0.f,0.f,0.f,0.f,0.f};

  stage(0, jbeg);
  u16* myP = s_P[w];
  int buf = 0;
  for (int j0 = jbeg; j0 < jend; j0 += 32, buf ^= 1) {
    wait_async0();            // my async loads into `buf` complete
    __syncthreads();          // everyone's staged data visible; prev reads done
    if (j0 + 32 < jend) stage(buf ^ 1, j0 + 32);

    const u16* kl = s_keys[buf];
    const u16* ml = s_msg[buf];
#pragma unroll
    for (int t = 0; t < 2; ++t) {
      int cb = j0 + t * 16;
      const u16* krow = kl + (t * 16 + ln) * 72;
      v8f s = {0.f,0.f,0.f,0.f,0.f,0.f,0.f,0.f};
      s = wmma_bf16(qf0, load_bfrag(krow + hs * 16), s);
      s = wmma_bf16(qf1, load_bfrag(krow + 32 + hs * 16), s);
      int c = cb + ln;
#pragma unroll
      for (int v = 0; v < 8; ++v) {
        int r = rb + v + 8 * hs;
        float sc = s[v] * 0.125f;                      // 1/sqrt(64)
        sc = fminf(20.f, fmaxf(-20.f, sc));
        float p = (r == c) ? 0.f : __expf(sc);         // diag mask: exp(-1e8)=0
        rs[v] += p;
        attn[(size_t)r * NA + c] = p;                  // unnormalized; K3 scales
        myP[(v + 8 * hs) * 40 + t * 16 + ln] = f2bf(p);
      }
    }
    asm volatile("s_wait_dscnt 0" ::: "memory");
    // re-read P as A-fragment (16x32)
    const u16* myRow = myP + ln * 40;
    uint4 plo = *reinterpret_cast<const uint4*>(myRow + hs * 16);
    uint4 phi = *reinterpret_cast<const uint4*>(myRow + hs * 16 + 8);
    B256 pu; pu.lo = plo; pu.hi = phi;
    v16bf pf = __builtin_bit_cast(v16bf, pu);
#pragma unroll
    for (int nt = 0; nt < 8; ++nt) {
      v16bf mf = load_bfrag(ml + (nt * 16 + ln) * 40 + hs * 16);
      accA[nt] = wmma_bf16(pf, mf, accA[nt]);
    }
  }

  // reduce partial row sums across the 16 lanes sharing hs
#pragma unroll
  for (int v = 0; v < 8; ++v)
    for (int m = 1; m < 16; m <<= 1) rs[v] += __shfl_xor(rs[v], m, 32);
  if (ln == 0)
#pragma unroll
    for (int v = 0; v < 8; ++v) rowsumP[(size_t)chunk * NA + rb + v + 8 * hs] = rs[v];

  // unnormalized partial attended for this chunk
  float* ap = attP + (size_t)chunk * NA * MSGD;
#pragma unroll
  for (int nt = 0; nt < 8; ++nt)
#pragma unroll
    for (int v = 0; v < 8; ++v) {
      int r = rb + v + 8 * hs, c = nt * 16 + ln;
      ap[(size_t)r * MSGD + c] = accA[nt][v];
    }
}

// ---------------------------------------------------------------------------
// K2b: combine chunk partials -> rinv + normalized attended (bf16)
__global__ __launch_bounds__(256) void combine_kernel(
    const float* __restrict__ attP, const float* __restrict__ rowsumP,
    u16* __restrict__ attB, float* __restrict__ rinv) {
  int idx = blockIdx.x * 256 + threadIdx.x;    // < NA*MSGD
  int r = idx >> 7, c = idx & 127;
  float s = 0.f, rsum = 0.f;
#pragma unroll
  for (int k = 0; k < NCHUNK; ++k) {
    s    += attP[(size_t)k * NA * MSGD + idx];
    rsum += rowsumP[(size_t)k * NA + r];
  }
  float inv = 1.f / rsum;
  attB[idx] = f2bf(s * inv);
  if (c == 0) rinv[r] = inv;
}

// ---------------------------------------------------------------------------
// K3: attn[i,j] *= rinv[i]   (float4 vectorized)
__global__ __launch_bounds__(256) void attn_norm_kernel(float* __restrict__ attn,
                                                        const float* __restrict__ rinv) {
  size_t idx = (size_t)blockIdx.x * 256 + threadIdx.x;     // one float4 each
  int row = (int)(idx >> 11);                               // (idx*4)/8192
  float s = rinv[row];
  float4* A = reinterpret_cast<float4*>(attn);
  float4 v = A[idx];
  v.x *= s; v.y *= s; v.z *= s; v.w *= s;
  A[idx] = v;
}

// ---------------------------------------------------------------------------
// K4: GRU. Each wave: 16x16 tile of h_new. Six GEMMs via WMMA + gate math.
__global__ __launch_bounds__(256) void gru_kernel(
    const u16* __restrict__ attB, const u16* __restrict__ hb, const float* __restrict__ h,
    const u16* __restrict__ WiTr, const u16* __restrict__ WiTz, const u16* __restrict__ WiTn,
    const u16* __restrict__ WhTr, const u16* __restrict__ WhTz, const u16* __restrict__ WhTn,
    const float* __restrict__ bi_r, const float* __restrict__ bi_z, const float* __restrict__ bi_n,
    const float* __restrict__ bh_n, float* __restrict__ h_new) {
  int g = blockIdx.x * 8 + (threadIdx.x >> 5);
  int lane = threadIdx.x & 31, ln = lane & 15, hs = lane >> 4;
  int rb = (g >> 4) * 16, n0 = (g & 15) * 16;

  v8f cr = {0.f,0.f,0.f,0.f,0.f,0.f,0.f,0.f};
  v8f cz = cr, cn = cr, hn = cr;

  const u16* arow = attB + (size_t)(rb + ln) * MSGD;
#pragma unroll
  for (int k0 = 0; k0 < MSGD; k0 += 32) {                    // input-side GEMMs
    v16bf a = load_afrag(arow, k0, hs);
    cr = wmma_bf16(a, load_bfrag(WiTr + (size_t)(n0 + ln) * MSGD + k0 + hs * 16), cr);
    cz = wmma_bf16(a, load_bfrag(WiTz + (size_t)(n0 + ln) * MSGD + k0 + hs * 16), cz);
    cn = wmma_bf16(a, load_bfrag(WiTn + (size_t)(n0 + ln) * MSGD + k0 + hs * 16), cn);
  }
  const u16* hrow = hb + (size_t)(rb + ln) * HID;
#pragma unroll
  for (int k0 = 0; k0 < HID; k0 += 32) {                     // hidden-side GEMMs
    v16bf a = load_afrag(hrow, k0, hs);
    cr = wmma_bf16(a, load_bfrag(WhTr + (size_t)(n0 + ln) * HID + k0 + hs * 16), cr);
    cz = wmma_bf16(a, load_bfrag(WhTz + (size_t)(n0 + ln) * HID + k0 + hs * 16), cz);
    hn = wmma_bf16(a, load_bfrag(WhTn + (size_t)(n0 + ln) * HID + k0 + hs * 16), hn);
  }

  int c = n0 + ln;
  float br = bi_r[c], bz = bi_z[c], bn = bi_n[c], bhn = bh_n[c];
#pragma unroll
  for (int v = 0; v < 8; ++v) {
    int r = rb + v + 8 * hs;
    float rg = 1.f / (1.f + __expf(-(cr[v] + br)));
    float zg = 1.f / (1.f + __expf(-(cz[v] + bz)));
    float nd = tanhf(cn[v] + bn + rg * (hn[v] + bhn));
    float hv = h[(size_t)r * HID + c];
    h_new[(size_t)r * HID + c] = (1.f - zg) * nd + zg * hv;
  }
}

// ---------------------------------------------------------------------------
extern "C" void kernel_launch(void* const* d_in, const int* in_sizes, int n_in,
                              void* d_out, int out_size, void* d_ws, size_t ws_size,
                              hipStream_t stream) {
  const float* h     = (const float*)d_in[0];
  const float* W_msg = (const float*)d_in[1];
  const float* b_msg = (const float*)d_in[2];
  const float* W_key = (const float*)d_in[3];
  const float* b_key = (const float*)d_in[4];
  const float* W_qry = (const float*)d_in[5];
  const float* b_qry = (const float*)d_in[6];
  const float* Wi_r  = (const float*)d_in[7];
  const float* bi_r  = (const float*)d_in[8];
  const float* Wi_z  = (const float*)d_in[9];
  const float* bi_z  = (const float*)d_in[10];
  const float* Wi_n  = (const float*)d_in[11];
  const float* bi_n  = (const float*)d_in[12];
  const float* Wh_r  = (const float*)d_in[13];
  const float* Wh_z  = (const float*)d_in[14];
  const float* Wh_n  = (const float*)d_in[15];
  const float* bh_n  = (const float*)d_in[16];

  float* h_new = (float*)d_out;
  float* attn  = (float*)d_out + (size_t)NA * HID;

  // workspace carve-up (256B aligned)
  char* base = (char*)d_ws;
  size_t o = 0;
  auto carve = [&](size_t bytes) { char* p = base + o; o += (bytes + 255) & ~(size_t)255; return p; };
  u16* hb    = (u16*)carve((size_t)NA * HID * 2);
  u16* MsgT  = (u16*)carve((size_t)MSGD * NA * 2);
  u16* keysB = (u16*)carve((size_t)NA * KEYD * 2);
  u16* qryB  = (u16*)carve((size_t)NA * KEYD * 2);
  u16* attB  = (u16*)carve((size_t)NA * MSGD * 2);
  u16* WmsgT = (u16*)carve((size_t)MSGD * HID * 2);
  u16* WkeyT = (u16*)carve((size_t)KEYD * HID * 2);
  u16* WqryT = (u16*)carve((size_t)KEYD * HID * 2);
  u16* WiTr  = (u16*)carve((size_t)HID * MSGD * 2);
  u16* WiTz  = (u16*)carve((size_t)HID * MSGD * 2);
  u16* WiTn  = (u16*)carve((size_t)HID * MSGD * 2);
  u16* WhTr  = (u16*)carve((size_t)HID * HID * 2);
  u16* WhTz  = (u16*)carve((size_t)HID * HID * 2);
  u16* WhTn  = (u16*)carve((size_t)HID * HID * 2);
  float* rinv    = (float*)carve((size_t)NA * 4);
  float* attP    = (float*)carve((size_t)NCHUNK * NA * MSGD * 4);   // 16 MB partials
  float* rowsumP = (float*)carve((size_t)NCHUNK * NA * 4);
  (void)ws_size; (void)in_sizes; (void)n_in; (void)out_size;

  // K0: converts + transposes
  cvt_kernel<<<(NA * HID + 255) / 256, 256, 0, stream>>>(h, hb, NA * HID);
  tcvt_kernel<<<(HID * MSGD + 255) / 256, 256, 0, stream>>>(W_msg, WmsgT, HID, MSGD);
  tcvt_kernel<<<(HID * KEYD + 255) / 256, 256, 0, stream>>>(W_key, WkeyT, HID, KEYD);
  tcvt_kernel<<<(HID * KEYD + 255) / 256, 256, 0, stream>>>(W_qry, WqryT, HID, KEYD);
  tcvt_kernel<<<(MSGD * HID + 255) / 256, 256, 0, stream>>>(Wi_r, WiTr, MSGD, HID);
  tcvt_kernel<<<(MSGD * HID + 255) / 256, 256, 0, stream>>>(Wi_z, WiTz, MSGD, HID);
  tcvt_kernel<<<(MSGD * HID + 255) / 256, 256, 0, stream>>>(Wi_n, WiTn, MSGD, HID);
  tcvt_kernel<<<(HID * HID + 255) / 256, 256, 0, stream>>>(Wh_r, WhTr, HID, HID);
  tcvt_kernel<<<(HID * HID + 255) / 256, 256, 0, stream>>>(Wh_z, WhTz, HID, HID);
  tcvt_kernel<<<(HID * HID + 255) / 256, 256, 0, stream>>>(Wh_n, WhTn, HID, HID);

  // K1: projections (8192 waves, 8 waves/block)
  proj_kernel<<<(NA / 16) * 16 / 8, 256, 0, stream>>>(hb, WmsgT, WkeyT, WqryT,
                                                      b_msg, b_key, b_qry,
                                                      MsgT, keysB, qryB);
  // K2: attention — grid (64 row-groups, 4 column chunks), 8 waves/block
  attn_kernel<<<dim3(NA / (16 * 8), NCHUNK), 256, 0, stream>>>(
      qryB, keysB, MsgT, attn, attP, rowsumP);
  // K2b: combine partials -> rinv + attB
  combine_kernel<<<(NA * MSGD) / 256, 256, 0, stream>>>(attP, rowsumP, attB, rinv);
  // K3: scale attn rows by 1/rowsum
  attn_norm_kernel<<<(int)(((size_t)NA * NA / 4) / 256), 256, 0, stream>>>(attn, rinv);
  // K4: GRU update (8192 waves)
  gru_kernel<<<(NA / 16) * (HID / 16) / 8, 256, 0, stream>>>(
      attB, hb, h, WiTr, WiTz, WiTn, WhTr, WhTz, WhTn,
      bi_r, bi_z, bi_n, bh_n, h_new);
}
